// MultiHeadAttention_49331994361960
// MI455X (gfx1250) — compile-verified, matrix operand order checked by reading, not run on hardware
//
#include <hip/hip_runtime.h>

// ---------------- problem constants (fixed by reference) ----------------
#define B_   2
#define S_   2048
#define DM   1024          // D_MODEL = H_ * DH
#define H_   16
#define DH   64
#define HB_  (H_ * B_)     // 32
#define MROWS (B_ * S_)    // 4096
#define MASK_VAL (-4294967295.0f)   // float(-2**32 + 1)
#define SCALE 0.125f                // 1/sqrt(64)

typedef __attribute__((ext_vector_type(16))) __bf16 v16bf;
typedef __attribute__((ext_vector_type(4)))  __bf16 v4bf;
typedef __attribute__((ext_vector_type(8)))  float  v8f;
typedef __attribute__((ext_vector_type(4)))  float  v4f;

static __device__ __forceinline__ v8f wmma_bf16(v16bf a, v16bf b, v8f c) {
    // D = A(16x32 bf16) * B(32x16 bf16) + C(16x16 f32)
    return __builtin_amdgcn_wmma_f32_16x16x32_bf16(
        /*neg_a=*/false, a, /*neg_b=*/false, b,
        /*c_mod=*/(short)0, c, /*reuse_a=*/false, /*reuse_b=*/false);
}

// locality 3 -> near (WGP) scope prefetch: pulls into all cache levels
#define PF(p) __builtin_prefetch((p), 0, 3)

// ---------------- precision conversion kernels ----------------
__global__ __launch_bounds__(256) void cast_f32_bf16(
    const float* __restrict__ in, __bf16* __restrict__ out, int n4) {
    int i = blockIdx.x * blockDim.x + threadIdx.x;
    int stride = gridDim.x * blockDim.x;
    const v4f* in4 = (const v4f*)in;
    v4bf* out4 = (v4bf*)out;
    for (; i < n4; i += stride) {
        v4f f = in4[i];
        v4bf o;
        o[0] = (__bf16)f[0]; o[1] = (__bf16)f[1];
        o[2] = (__bf16)f[2]; o[3] = (__bf16)f[3];
        out4[i] = o;
    }
}

// Wt[n*K + k] = (bf16) W[k*N + n]  -- gives contiguous-K B-fragment loads
__global__ __launch_bounds__(256) void transpose_cast(
    const float* __restrict__ W, __bf16* __restrict__ Wt, int K, int N) {
    __shared__ float tile[32][33];
    int k0 = blockIdx.x * 32, n0 = blockIdx.y * 32;
    int tx = threadIdx.x & 31, ty = threadIdx.x >> 5;   // ty in 0..7
    for (int i = ty; i < 32; i += 8)
        tile[i][tx] = W[(size_t)(k0 + i) * N + n0 + tx];
    __syncthreads();
    for (int i = ty; i < 32; i += 8)
        Wt[(size_t)(n0 + i) * K + k0 + tx] = (__bf16)tile[tx][i];
}

// ---------------- generic WMMA GEMM:  C[M,N] = A[M,K] * Bt[N,K]^T + bias ----
// Per-wave tile: 16 (M) x 64 (N) -> 4 accumulators reusing one A fragment.
// mode 0: store bf16, head layout [h*B+b][s][64]            (Q, K)
// mode 1: store bf16, transposed head layout [h*B+b][64][s] (V)
// mode 2: store f32 row-major [M][N] (+bias), non-temporal  (final projection)
__global__ __launch_bounds__(128) void gemm_bf16_wmma(
    const __bf16* __restrict__ A, const __bf16* __restrict__ Bt,
    const float* __restrict__ bias,
    __bf16* __restrict__ outHead, float* __restrict__ outF32,
    int M, int N, int K, int mode) {
    int wave = threadIdx.x >> 5, lane = threadIdx.x & 31;
    int tile = blockIdx.x * 4 + wave;
    int ntN = N >> 6;                 // 64-wide N tiles
    int m0 = (tile / ntN) << 4;
    int n0 = (tile % ntN) << 6;
    if (m0 >= M) return;

    int row = lane & 15;              // A row / B column within a 16-wide tile
    int hi  = lane >> 4;              // which K-half this lane holds
    int kh  = hi << 4;

    const __bf16* Ap  = A  + (size_t)(m0 + row) * K + kh;
    const __bf16* Bp0 = Bt + (size_t)(n0 +  0 + row) * K + kh;
    const __bf16* Bp1 = Bt + (size_t)(n0 + 16 + row) * K + kh;
    const __bf16* Bp2 = Bt + (size_t)(n0 + 32 + row) * K + kh;
    const __bf16* Bp3 = Bt + (size_t)(n0 + 48 + row) * K + kh;

    v8f acc0 = {}, acc1 = {}, acc2 = {}, acc3 = {};
#pragma unroll 2
    for (int k0 = 0; k0 < K; k0 += 32) {
        // cover L2->WGP latency: prefetch ~8 k-steps ahead on all 5 streams
        PF(Ap  + k0 + 256);
        PF(Bp0 + k0 + 256);
        PF(Bp1 + k0 + 256);
        PF(Bp2 + k0 + 256);
        PF(Bp3 + k0 + 256);
        v16bf a  = *(const v16bf*)(Ap  + k0);
        v16bf b0 = *(const v16bf*)(Bp0 + k0);
        v16bf b1 = *(const v16bf*)(Bp1 + k0);
        v16bf b2 = *(const v16bf*)(Bp2 + k0);
        v16bf b3 = *(const v16bf*)(Bp3 + k0);
        acc0 = wmma_bf16(a, b0, acc0);
        acc1 = wmma_bf16(a, b1, acc1);
        acc2 = wmma_bf16(a, b2, acc2);
        acc3 = wmma_bf16(a, b3, acc3);
    }

    v8f accs[4] = {acc0, acc1, acc2, acc3};
#pragma unroll
    for (int t = 0; t < 4; ++t) {
        int nn = n0 + t * 16 + row;
        float bval = bias ? bias[nn] : 0.0f;
#pragma unroll
        for (int r = 0; r < 8; ++r) {
            int mr = m0 + r + 8 * hi;           // C layout: VGPR r -> M = r (+8 hi)
            float v = accs[t][r] + bval;
            if (mode == 2) {
                __builtin_nontemporal_store(v, &outF32[(size_t)mr * N + nn]);
            } else {
                int bb = mr / S_, ss = mr % S_; // mr = b*S + s
                int hh = nn >> 6, dd = nn & 63; // nn = h*64 + d
                size_t hb = (size_t)hh * B_ + bb;
                size_t idx = (mode == 0)
                    ? (hb * S_ + ss) * DH + dd
                    : (hb * DH + dd) * (size_t)S_ + ss;
                outHead[idx] = (__bf16)v;
            }
        }
    }
}

// ---------------- flash attention + raw score streaming ----------------
// Q,K: [HB][S][64] bf16   Vt: [HB][64][S] bf16
// raw: [HB][S][S] f32 (pre-mask, scaled, NT-stored)   ctx: [B][S][H*64] bf16
__global__ __launch_bounds__(128) void attention_kernel(
    const __bf16* __restrict__ Qh, const __bf16* __restrict__ Kh,
    const __bf16* __restrict__ Vth,
    float* __restrict__ raw, __bf16* __restrict__ ctx) {
    __shared__ __align__(64) __bf16 lp[4][16 * 32];   // per-wave P transpose buffer

    int wave = threadIdx.x >> 5, lane = threadIdx.x & 31;
    int hb = blockIdx.y;
    int q0 = blockIdx.x * 64 + wave * 16;
    int h = hb / B_, b = hb % B_;
    int row = lane & 15, hi = lane >> 4, kh = hi << 4, col = lane & 15;

    const __bf16* Q  = Qh  + (size_t)hb * S_ * DH;
    const __bf16* Kp = Kh  + (size_t)hb * S_ * DH;
    const __bf16* Vt = Vth + (size_t)hb * DH * S_;
    float* rawp = raw + (size_t)hb * S_ * S_;

    // Q fragments for this 16-query tile (dims 0..31 and 32..63), loaded once
    v16bf aq0 = *(const v16bf*)(Q + (size_t)(q0 + row) * DH + kh);
    v16bf aq1 = *(const v16bf*)(Q + (size_t)(q0 + row) * DH + 32 + kh);

    float m[8], l[8];
#pragma unroll
    for (int r = 0; r < 8; ++r) { m[r] = -1e30f; l[r] = 0.0f; }
    v8f acc0 = {}, acc1 = {}, acc2 = {}, acc3 = {};

    for (int kc = 0; kc < S_; kc += 32) {
        // prefetch next key chunk (K rows and V columns) into WGP cache
        PF(Kp + (size_t)(kc + 32 + row) * DH + kh);
        PF(Kp + (size_t)(kc + 48 + row) * DH + kh);
        PF(Vt + (size_t)row * S_ + kc + 32 + kh);
        PF(Vt + (size_t)(48 + row) * S_ + kc + 32 + kh);

        // ---- scores: two 16x16 tiles covering keys kc..kc+31
        v16bf bkA = *(const v16bf*)(Kp + (size_t)(kc + row) * DH + kh);
        v16bf bkB = *(const v16bf*)(Kp + (size_t)(kc + row) * DH + 32 + kh);
        v8f sc0 = {};
        sc0 = wmma_bf16(aq0, bkA, sc0);
        sc0 = wmma_bf16(aq1, bkB, sc0);
        v16bf bkC = *(const v16bf*)(Kp + (size_t)(kc + 16 + row) * DH + kh);
        v16bf bkD = *(const v16bf*)(Kp + (size_t)(kc + 16 + row) * DH + 32 + kh);
        v8f sc1 = {};
        sc1 = wmma_bf16(aq0, bkC, sc1);
        sc1 = wmma_bf16(aq1, bkD, sc1);

        // ---- scale, stream raw scores (non-temporal), mask, online softmax
#pragma unroll
        for (int r = 0; r < 8; ++r) {
            int qrow = q0 + r + 8 * hi;
            float s0 = sc0[r] * SCALE;
            float s1 = sc1[r] * SCALE;
            __builtin_nontemporal_store(s0, &rawp[(size_t)qrow * S_ + kc + col]);
            __builtin_nontemporal_store(s1, &rawp[(size_t)qrow * S_ + kc + 16 + col]);
            if (kc + col > qrow)      s0 = MASK_VAL;        // causal
            if (kc + 16 + col > qrow) s1 = MASK_VAL;

            float mt = fmaxf(s0, s1);                       // row max over 16 lanes
            mt = fmaxf(mt, __shfl_xor(mt, 1));
            mt = fmaxf(mt, __shfl_xor(mt, 2));
            mt = fmaxf(mt, __shfl_xor(mt, 4));
            mt = fmaxf(mt, __shfl_xor(mt, 8));
            float newm = fmaxf(m[r], mt);
            float corr = __expf(m[r] - newm);
            m[r] = newm;
            float p0 = __expf(s0 - newm);
            float p1 = __expf(s1 - newm);
            float ps = p0 + p1;                             // row sum
            ps += __shfl_xor(ps, 1);
            ps += __shfl_xor(ps, 2);
            ps += __shfl_xor(ps, 4);
            ps += __shfl_xor(ps, 8);
            l[r] = l[r] * corr + ps;
            acc0[r] *= corr; acc1[r] *= corr;
            acc2[r] *= corr; acc3[r] *= corr;
            // C-layout -> LDS (A-layout read below); DS is in-order per wave
            lp[wave][(r + 8 * hi) * 32 + col]      = (__bf16)p0;
            lp[wave][(r + 8 * hi) * 32 + 16 + col] = (__bf16)p1;
        }

        // ---- P(16x32) * V(32x64): A from LDS, B fragments contiguous from Vt
        v16bf ap = *(const v16bf*)(&lp[wave][row * 32 + kh]);
        v16bf bv0 = *(const v16bf*)(Vt + (size_t)( 0 + row) * S_ + kc + kh);
        v16bf bv1 = *(const v16bf*)(Vt + (size_t)(16 + row) * S_ + kc + kh);
        v16bf bv2 = *(const v16bf*)(Vt + (size_t)(32 + row) * S_ + kc + kh);
        v16bf bv3 = *(const v16bf*)(Vt + (size_t)(48 + row) * S_ + kc + kh);
        acc0 = wmma_bf16(ap, bv0, acc0);
        acc1 = wmma_bf16(ap, bv1, acc1);
        acc2 = wmma_bf16(ap, bv2, acc2);
        acc3 = wmma_bf16(ap, bv3, acc3);
    }

    // ---- normalize and emit heads in [B][S][H*64] bf16 for the out-projection
#pragma unroll
    for (int r = 0; r < 8; ++r) {
        int ss = q0 + r + 8 * hi;
        float inv = 1.0f / l[r];
        size_t base = ((size_t)b * S_ + ss) * DM + (size_t)h * DH;
        ctx[base +  0 + col] = (__bf16)(acc0[r] * inv);
        ctx[base + 16 + col] = (__bf16)(acc1[r] * inv);
        ctx[base + 32 + col] = (__bf16)(acc2[r] * inv);
        ctx[base + 48 + col] = (__bf16)(acc3[r] * inv);
    }
}

// ---------------- launch ----------------
extern "C" void kernel_launch(void* const* d_in, const int* in_sizes, int n_in,
                              void* d_out, int out_size, void* d_ws, size_t ws_size,
                              hipStream_t stream) {
    const float* x  = (const float*)d_in[0];
    const float* Wq = (const float*)d_in[1];
    const float* bq = (const float*)d_in[2];
    const float* Wk = (const float*)d_in[3];
    const float* bk = (const float*)d_in[4];
    const float* Wv = (const float*)d_in[5];
    const float* bv = (const float*)d_in[6];
    const float* Wo = (const float*)d_in[7];
    const float* bo = (const float*)d_in[8];

    float* out = (float*)d_out;                       // [B][S][DM]
    float* raw = out + (size_t)B_ * S_ * DM;          // [HB][S][S]

    // workspace carve (48 MB total, all bf16)
    char* w = (char*)d_ws;
    __bf16* xb  = (__bf16*)(w + (0ull  << 20));   // 8 MB  x as bf16 [M][K]
    __bf16* Wqt = (__bf16*)(w + (8ull  << 20));   // 2 MB  [N][K]
    __bf16* Wkt = (__bf16*)(w + (10ull << 20));   // 2 MB
    __bf16* Wvt = (__bf16*)(w + (12ull << 20));   // 2 MB
    __bf16* Wot = (__bf16*)(w + (14ull << 20));   // 2 MB
    __bf16* Qh  = (__bf16*)(w + (16ull << 20));   // 8 MB  [HB][S][64]
    __bf16* Kh  = (__bf16*)(w + (24ull << 20));   // 8 MB  [HB][S][64]
    __bf16* Vth = (__bf16*)(w + (32ull << 20));   // 8 MB  [HB][64][S]
    __bf16* ctx = (__bf16*)(w + (40ull << 20));   // 8 MB  [B][S][DM]

    cast_f32_bf16<<<1024, 256, 0, stream>>>(x, xb, MROWS * DM / 4);
    transpose_cast<<<dim3(DM / 32, DM / 32), 256, 0, stream>>>(Wq, Wqt, DM, DM);
    transpose_cast<<<dim3(DM / 32, DM / 32), 256, 0, stream>>>(Wk, Wkt, DM, DM);
    transpose_cast<<<dim3(DM / 32, DM / 32), 256, 0, stream>>>(Wv, Wvt, DM, DM);
    transpose_cast<<<dim3(DM / 32, DM / 32), 256, 0, stream>>>(Wo, Wot, DM, DM);

    int tiles = (MROWS / 16) * (DM / 64);             // 4096 wave-tiles (16x64 each)
    gemm_bf16_wmma<<<tiles / 4, 128, 0, stream>>>(xb, Wqt, bq, Qh,  nullptr, MROWS, DM, DM, 0);
    gemm_bf16_wmma<<<tiles / 4, 128, 0, stream>>>(xb, Wkt, bk, Kh,  nullptr, MROWS, DM, DM, 0);
    gemm_bf16_wmma<<<tiles / 4, 128, 0, stream>>>(xb, Wvt, bv, Vth, nullptr, MROWS, DM, DM, 1);

    attention_kernel<<<dim3(S_ / 64, HB_), 128, 0, stream>>>(Qh, Kh, Vth, raw, ctx);

    gemm_bf16_wmma<<<tiles / 4, 128, 0, stream>>>(ctx, Wot, bo, nullptr, out, MROWS, DM, DM, 2);
}
